// GruPredictor_86139864089184
// MI455X (gfx1250) — matrix operand order repaired and checked by reading, not compiled, not run on hardware
//
#include <hip/hip_runtime.h>

typedef __attribute__((ext_vector_type(16))) __bf16 v16bf;
typedef __attribute__((ext_vector_type(8)))  float  v8f;

#define Bsz 256
#define Tsz 512
#define Dx  64
#define Hd  512
#define KCAT 576         // D + H
#define NKT_CAT 18       // 576 / 32
#define NKT_H   16       // 512 / 32
#define FSEQ 32

// ---------------------------------------------------------------------------
// Load a 16-wide bf16 A-fragment slice for this lane from an f32 row.
// A 16x32 bf16 layout used here (contiguous k, same as B): lane(=16*hi+m) row m holds
//   K = kbase + {0..15}, where kbase already includes hi*16.
// ---------------------------------------------------------------------------
__device__ __forceinline__ v16bf load_a_frag_f32(const float* __restrict__ row, int kbase) {
  const float4* p0 = reinterpret_cast<const float4*>(row + kbase);
  const float4* p1 = reinterpret_cast<const float4*>(row + kbase + 8);
  float4 a0 = p0[0], a1 = p0[1], a2 = p1[0], a3 = p1[1];
  v16bf r;
  r[0]  = (__bf16)a0.x; r[1]  = (__bf16)a0.y; r[2]  = (__bf16)a0.z; r[3]  = (__bf16)a0.w;
  r[4]  = (__bf16)a1.x; r[5]  = (__bf16)a1.y; r[6]  = (__bf16)a1.z; r[7]  = (__bf16)a1.w;
  r[8]  = (__bf16)a2.x; r[9]  = (__bf16)a2.y; r[10] = (__bf16)a2.z; r[11] = (__bf16)a2.w;
  r[12] = (__bf16)a3.x; r[13] = (__bf16)a3.y; r[14] = (__bf16)a3.z; r[15] = (__bf16)a3.w;
  return r;
}

__device__ __forceinline__ float sigmoidf_(float x) {
  return 1.0f / (1.0f + __expf(-x));
}

// ---------------------------------------------------------------------------
// Weight packing into B-fragment blocked layout:
//   out[(ct*NKT + kt)*32 + lane][e] = Bmat(k = kt*32 + hi*16 + e, n = ct*16 + m)
// B 32x16 bf16 layout used here (contiguous k, same as A): lane(=16*hi+m) holds column n, K = hi*16 + e.
// ---------------------------------------------------------------------------
__global__ void pack_wcat_kernel(const float* __restrict__ W_ih,   // [1536, 64]
                                 const float* __restrict__ W_hh,   // [1536, 512]
                                 __bf16* __restrict__ out) {
  int tid = blockIdx.x * blockDim.x + threadIdx.x;
  const int total = 96 * NKT_CAT * 32;                 // colTiles * kTiles * lanes
  if (tid >= total) return;
  int lane = tid & 31;
  int kt   = (tid >> 5) % NKT_CAT;
  int ct   = tid / (NKT_CAT * 32);
  int n    = ct * 16 + (lane & 15);
  int kb   = kt * 32 + ((lane >> 4) ? 16 : 0);
  __bf16* dst = out + (size_t)tid * 16;
#pragma unroll
  for (int e = 0; e < 16; ++e) {
    int k = kb + e;
    float v = (k < Dx) ? W_ih[n * Dx + k] : W_hh[n * Hd + (k - Dx)];
    dst[e] = (__bf16)v;
  }
}

__global__ void pack_rowmajor_kernel(const float* __restrict__ W,  // [N, Kdim] row major
                                     __bf16* __restrict__ out,
                                     int nColTiles, int Kdim) {
  int tid = blockIdx.x * blockDim.x + threadIdx.x;
  int nkt = Kdim >> 5;
  int total = nColTiles * nkt * 32;
  if (tid >= total) return;
  int lane = tid & 31;
  int kt   = (tid >> 5) % nkt;
  int ct   = tid / (nkt * 32);
  int n    = ct * 16 + (lane & 15);
  int kb   = kt * 32 + ((lane >> 4) ? 16 : 0);
  __bf16* dst = out + (size_t)tid * 16;
#pragma unroll
  for (int e = 0; e < 16; ++e) dst[e] = (__bf16)W[n * Kdim + kb + e];
}

__global__ void zero_kernel(float* __restrict__ p, int n) {
  int i = blockIdx.x * blockDim.x + threadIdx.x;
  if (i < n) p[i] = 0.0f;
}

// Record decoder outputs: s_seq[b, t, :] = s ; h_seq[b, t, :] = h (pre-update)
__global__ void copy_out_kernel(const float* __restrict__ s,
                                const float* __restrict__ h,
                                float* __restrict__ out, int t) {
  int tid = blockIdx.x * blockDim.x + threadIdx.x;
  const int total = Bsz * (Dx + Hd);
  if (tid >= total) return;
  int b = tid / (Dx + Hd);
  int i = tid - b * (Dx + Hd);
  if (i < Dx) {
    out[((size_t)b * FSEQ + t) * Dx + i] = s[b * Dx + i];
  } else {
    int j = i - Dx;
    out[(size_t)Bsz * FSEQ * Dx + ((size_t)b * FSEQ + t) * Hd + j] = h[b * Hd + j];
  }
}

// ---------------------------------------------------------------------------
// One fused GRU cell step: h_out = GRUCell([x | h_in], h_in)
// Per wave: one 16(row) x 16(col j) tile; accumulators for r, z, i_n, h_n.
// Grid: (8 colGroups, 16 rowTiles), block = 128 (4 waves -> colTiles 0..31).
// ---------------------------------------------------------------------------
__global__ __launch_bounds__(128) void gru_step_kernel(
    const float* __restrict__ x_base, int x_row_stride,   // x row b at x_base + b*stride
    const float* __restrict__ h_in,                       // [256, 512]
    float* __restrict__ h_out,                            // [256, 512]
    const __bf16* __restrict__ wcat,                      // packed [96][18][32][16]
    const float* __restrict__ b_ih,                       // [1536]
    const float* __restrict__ b_hh) {                     // [1536]
  const int lane = threadIdx.x & 31;
  const int wave = threadIdx.x >> 5;
  const int rowTile = blockIdx.y;                 // 0..15
  const int colTile = blockIdx.x * 4 + wave;      // 0..31
  const int rowBase = rowTile * 16;
  const int hi = lane >> 4;
  const int m  = lane & 15;
  const int aRow = rowBase + m;

  const float* xrow = x_base + (size_t)aRow * x_row_stride;
  const float* hrow = h_in + aRow * Hd;

  v8f acc_r = {}, acc_z = {}, acc_in = {}, acc_hn = {};

  const int ct_r = colTile;
  const int ct_z = 32 + colTile;
  const int ct_n = 64 + colTile;

  const __bf16* base_r = wcat + ((size_t)(ct_r * NKT_CAT) * 32 + lane) * 16;
  const __bf16* base_z = wcat + ((size_t)(ct_z * NKT_CAT) * 32 + lane) * 16;
  const __bf16* base_n = wcat + ((size_t)(ct_n * NKT_CAT) * 32 + lane) * 16;

  for (int kt = 0; kt < NKT_CAT; ++kt) {
    v16bf a;
    if (kt < 2) a = load_a_frag_f32(xrow, kt * 32 + hi * 16);          // x part (K < 64)
    else        a = load_a_frag_f32(hrow, kt * 32 - Dx + hi * 16);     // h part

    const v16bf fr = *reinterpret_cast<const v16bf*>(base_r + (size_t)kt * 32 * 16);
    const v16bf fz = *reinterpret_cast<const v16bf*>(base_z + (size_t)kt * 32 * 16);
    const v16bf fn = *reinterpret_cast<const v16bf*>(base_n + (size_t)kt * 32 * 16);

    if (kt + 1 < NKT_CAT) {   // prefetch next k-tile's B fragments
      __builtin_prefetch(base_r + (size_t)(kt + 1) * 32 * 16, 0, 3);
      __builtin_prefetch(base_z + (size_t)(kt + 1) * 32 * 16, 0, 3);
      __builtin_prefetch(base_n + (size_t)(kt + 1) * 32 * 16, 0, 3);
    }

    acc_r = __builtin_amdgcn_wmma_f32_16x16x32_bf16(false, a, false, fr, (short)0, acc_r, false, false);
    acc_z = __builtin_amdgcn_wmma_f32_16x16x32_bf16(false, a, false, fz, (short)0, acc_z, false, false);
    if (kt < 2)
      acc_in = __builtin_amdgcn_wmma_f32_16x16x32_bf16(false, a, false, fn, (short)0, acc_in, false, false);
    else
      acc_hn = __builtin_amdgcn_wmma_f32_16x16x32_bf16(false, a, false, fn, (short)0, acc_hn, false, false);
  }

  // Epilogue: gates + state update. C/D layout: element (v, lane) = row v+8*hi, col m.
  const int j = colTile * 16 + m;
  const float br  = b_ih[j]          + b_hh[j];
  const float bz  = b_ih[Hd + j]     + b_hh[Hd + j];
  const float bin = b_ih[2 * Hd + j];
  const float bhn = b_hh[2 * Hd + j];
#pragma unroll
  for (int v = 0; v < 8; ++v) {
    int row = rowBase + v + hi * 8;
    float r = sigmoidf_(acc_r[v] + br);
    float z = sigmoidf_(acc_z[v] + bz);
    float n = tanhf(acc_in[v] + bin + r * (acc_hn[v] + bhn));
    float hp = h_in[row * Hd + j];
    h_out[row * Hd + j] = (1.0f - z) * n + z * hp;
  }
}

// ---------------------------------------------------------------------------
// out[b, j] = act( in[b, :512] @ Wpk(:, j) + bias[j] ),  act = ReLU or identity
// ---------------------------------------------------------------------------
__global__ __launch_bounds__(128) void linear_kernel(
    const float* __restrict__ in,       // [256, 512]
    const __bf16* __restrict__ wpk,     // packed [nColTiles][16][32][16]
    const float* __restrict__ bias,
    float* __restrict__ out, int Nout, int relu) {
  const int lane = threadIdx.x & 31;
  const int wave = threadIdx.x >> 5;
  const int rowTile = blockIdx.y;
  const int colTile = blockIdx.x * 4 + wave;
  const int rowBase = rowTile * 16;
  const int hi = lane >> 4;
  const int m  = lane & 15;
  const float* arow = in + (size_t)(rowBase + m) * Hd;

  v8f acc = {};
  const __bf16* basew = wpk + ((size_t)(colTile * NKT_H) * 32 + lane) * 16;
  for (int kt = 0; kt < NKT_H; ++kt) {
    v16bf a = load_a_frag_f32(arow, kt * 32 + hi * 16);
    v16bf b = *reinterpret_cast<const v16bf*>(basew + (size_t)kt * 32 * 16);
    if (kt + 1 < NKT_H) __builtin_prefetch(basew + (size_t)(kt + 1) * 32 * 16, 0, 3);
    acc = __builtin_amdgcn_wmma_f32_16x16x32_bf16(false, a, false, b, (short)0, acc, false, false);
  }
  const int j = colTile * 16 + m;
  const float bb = bias[j];
#pragma unroll
  for (int v = 0; v < 8; ++v) {
    int row = rowBase + v + hi * 8;
    float o = acc[v] + bb;
    if (relu) o = fmaxf(o, 0.0f);
    out[(size_t)row * Nout + j] = o;
  }
}

// ---------------------------------------------------------------------------
extern "C" void kernel_launch(void* const* d_in, const int* in_sizes, int n_in,
                              void* d_out, int out_size, void* d_ws, size_t ws_size,
                              hipStream_t stream) {
  const float* state_seq = (const float*)d_in[0];   // [256, 512, 64]
  const float* W_ih  = (const float*)d_in[1];       // [1536, 64]
  const float* W_hh  = (const float*)d_in[2];       // [1536, 512]
  const float* b_ih  = (const float*)d_in[3];       // [1536]
  const float* b_hh  = (const float*)d_in[4];       // [1536]
  const float* fc1_w = (const float*)d_in[5];       // [512, 512]
  const float* fc1_b = (const float*)d_in[6];       // [512]
  const float* fc2_w = (const float*)d_in[7];       // [64, 512]
  const float* fc2_b = (const float*)d_in[8];       // [64]
  float* out = (float*)d_out;

  // Workspace carve-out (~3.8 MB total)
  char* ws = (char*)d_ws;
  size_t off = 0;
  auto carve = [&](size_t bytes) -> void* {
    void* p = ws + off;
    off += (bytes + 255) & ~(size_t)255;
    return p;
  };
  __bf16* wcat_pk = (__bf16*)carve((size_t)96 * NKT_CAT * 32 * 16 * 2);
  __bf16* fc1_pk  = (__bf16*)carve((size_t)32 * NKT_H  * 32 * 16 * 2);
  __bf16* fc2_pk  = (__bf16*)carve((size_t) 4 * NKT_H  * 32 * 16 * 2);
  float*  h0   = (float*)carve((size_t)Bsz * Hd * 4);
  float*  h1   = (float*)carve((size_t)Bsz * Hd * 4);
  float*  tmp  = (float*)carve((size_t)Bsz * Hd * 4);
  float*  sbuf = (float*)carve((size_t)Bsz * Dx * 4);

  // One-time prep (re-run every call; deterministic)
  { int tot = 96 * NKT_CAT * 32;
    pack_wcat_kernel<<<(tot + 255) / 256, 256, 0, stream>>>(W_ih, W_hh, wcat_pk); }
  { int tot = 32 * NKT_H * 32;
    pack_rowmajor_kernel<<<(tot + 255) / 256, 256, 0, stream>>>(fc1_w, fc1_pk, 32, Hd); }
  { int tot = 4 * NKT_H * 32;
    pack_rowmajor_kernel<<<(tot + 255) / 256, 256, 0, stream>>>(fc2_w, fc2_pk, 4, Hd); }
  { int n = Bsz * Hd;
    zero_kernel<<<(n + 255) / 256, 256, 0, stream>>>(h0, n); }

  const dim3 gGru(8, 16), blk(128);
  float* hc = h0;
  float* hn = h1;

  // Encoder: 512 GRU steps over state_seq
  for (int t = 0; t < Tsz; ++t) {
    gru_step_kernel<<<gGru, blk, 0, stream>>>(state_seq + (size_t)t * Dx, Tsz * Dx,
                                              hc, hn, wcat_pk, b_ih, b_hh);
    float* sw = hc; hc = hn; hn = sw;
  }

  // s0 = fc_predict(h_n)
  linear_kernel<<<dim3(8, 16), blk, 0, stream>>>(hc, fc1_pk, fc1_b, tmp, Hd, 1);
  linear_kernel<<<dim3(1, 16), blk, 0, stream>>>(tmp, fc2_pk, fc2_b, sbuf, Dx, 0);

  // Decoder rollout: record (s, h), then step
  for (int t = 0; t < FSEQ; ++t) {
    int tot = Bsz * (Dx + Hd);
    copy_out_kernel<<<(tot + 255) / 256, 256, 0, stream>>>(sbuf, hc, out, t);
    gru_step_kernel<<<gGru, blk, 0, stream>>>(sbuf, Dx, hc, hn, wcat_pk, b_ih, b_hh);
    float* sw = hc; hc = hn; hn = sw;
    linear_kernel<<<dim3(8, 16), blk, 0, stream>>>(hc, fc1_pk, fc1_b, tmp, Hd, 1);
    linear_kernel<<<dim3(1, 16), blk, 0, stream>>>(tmp, fc2_pk, fc2_b, sbuf, Dx, 0);
  }
}
